// MultiHeadAttention_32100585571131
// MI455X (gfx1250) — matrix operand order repaired and checked by reading, not compiled, not run on hardware
//
#include <hip/hip_runtime.h>
#include <hip/hip_bf16.h>

typedef unsigned short u16;
typedef __bf16 bf16_t;
typedef bf16_t bf16x8  __attribute__((ext_vector_type(8)));
typedef bf16_t bf16x16 __attribute__((ext_vector_type(16)));
typedef float  f32x8   __attribute__((ext_vector_type(8)));

#define B_SZ   2
#define S_SZ   2048
#define DM     2048
#define HQ_N   16
#define HKV_N  4
#define DK_N   128

__device__ __forceinline__ u16 f2bf(float f) {
    unsigned int u = __float_as_uint(f);
    unsigned int r = u + 0x7FFFu + ((u >> 16) & 1u);   // round-to-nearest-even
    return (u16)(r >> 16);
}

__device__ __forceinline__ bf16x16 make_frag(const u16* lo_p, const u16* hi_p) {
    bf16x8 lo = *(const bf16x8*)lo_p;
    bf16x8 hi = *(const bf16x8*)hi_p;
    bf16x16 r;
    #pragma unroll
    for (int i = 0; i < 8; ++i) { r[i] = lo[i]; r[i + 8] = hi[i]; }
    return r;
}

// ---------------------------------------------------------------- convert ----
__global__ void k_f32_to_bf16(const float* __restrict__ in, u16* __restrict__ out, int n) {
    for (int i = blockIdx.x * blockDim.x + threadIdx.x; i < n; i += gridDim.x * blockDim.x)
        out[i] = f2bf(in[i]);
}

// f32 [K][N] -> bf16 [N][K] tiled transpose (coalesced both ways through LDS)
__global__ __launch_bounds__(256) void k_f32_to_bf16_T(
    const float* __restrict__ in, u16* __restrict__ out, int K, int N)
{
    __shared__ float tile[32][33];
    const int tx = threadIdx.x, ty = threadIdx.y;        // 32 x 8
    const int n0 = blockIdx.x * 32, k0 = blockIdx.y * 32;
    #pragma unroll
    for (int i = 0; i < 4; ++i) {
        int k = k0 + ty + i * 8;
        tile[ty + i * 8][tx] = in[(size_t)k * N + n0 + tx];
    }
    __syncthreads();
    #pragma unroll
    for (int i = 0; i < 4; ++i) {
        int n = n0 + ty + i * 8;
        out[(size_t)n * K + k0 + tx] = f2bf(tile[tx][ty + i * 8]);
    }
}

// ------------------------------------------------------------------- GEMM ----
// C[M,N] (f32, +bias) = A_bf16[M,K] @ B, with B given pre-transposed as
// BT_bf16[N][K].  128 threads = 4 waves; block tile 128(M) x 64(N); each wave
// owns 32 rows (two 16-row groups) so every B-fragment feeds two WMMAs.
__global__ __launch_bounds__(128) void k_gemm_bf16(
    const u16* __restrict__ A, const u16* __restrict__ BT,
    float* __restrict__ C, const float* __restrict__ bias,
    int M, int N, int K)
{
    const int wave = threadIdx.x >> 5, lane = threadIdx.x & 31;
    const int hb = lane >> 4, l15 = lane & 15;
    const int rt = blockIdx.x * 128 + wave * 32, nb = blockIdx.y * 64;

    f32x8 acc[2][4];
    #pragma unroll
    for (int g = 0; g < 2; ++g)
        #pragma unroll
        for (int t = 0; t < 4; ++t)
            #pragma unroll
            for (int i = 0; i < 8; ++i) acc[g][t][i] = 0.0f;

    const u16* a0 = A + (size_t)(rt + l15) * K + hb * 8;        // A-layout: lane half -> K-base 0/8
    const u16* a1 = A + (size_t)(rt + 16 + l15) * K + hb * 8;
    const u16* bbase = BT + (size_t)(nb + l15) * K + hb * 8;    // B-layout: lane half -> K-base 0/8

    #pragma unroll 2
    for (int k0 = 0; k0 < K; k0 += 32) {
        bf16x16 af0 = make_frag(a0 + k0, a0 + k0 + 16);
        bf16x16 af1 = make_frag(a1 + k0, a1 + k0 + 16);
        #pragma unroll
        for (int nt = 0; nt < 4; ++nt) {
            const u16* bp = bbase + (size_t)(nt * 16) * K + k0;
            bf16x16 bf = make_frag(bp, bp + 16);
            acc[0][nt] = __builtin_amdgcn_wmma_f32_16x16x32_bf16(
                false, af0, false, bf, (short)0, acc[0][nt], false, false);
            acc[1][nt] = __builtin_amdgcn_wmma_f32_16x16x32_bf16(
                false, af1, false, bf, (short)0, acc[1][nt], false, false);
        }
    }

    #pragma unroll
    for (int g = 0; g < 2; ++g)
        #pragma unroll
        for (int nt = 0; nt < 4; ++nt) {
            int col = nb + nt * 16 + l15;
            float bv = bias ? bias[col] : 0.0f;
            #pragma unroll
            for (int i = 0; i < 8; ++i) {
                int r = rt + g * 16 + i + hb * 8;
                C[(size_t)r * N + col] = acc[g][nt][i] + bv;
            }
        }
}

// ------------------------------------------------------------------- RoPE ----
__global__ void k_rope_q(const float* __restrict__ q, const float* __restrict__ cosb,
                         const float* __restrict__ sinb, u16* __restrict__ out)
{
    int idx = blockIdx.x * blockDim.x + threadIdx.x;       // B*S*HQ*64 threads
    int j = idx & 63; int t = idx >> 6;
    int hq = t & 15;  t >>= 4;
    int s  = t & 2047; int b = t >> 11;
    size_t base = ((size_t)(b * S_SZ + s)) * DM + hq * DK_N + 2 * j;
    float tr = q[base], ti = q[base + 1];
    float c = cosb[s * 64 + j], sn = sinb[s * 64 + j];
    size_t ob = (((size_t)(b * HQ_N + hq)) * S_SZ + s) * DK_N + 2 * j;
    out[ob]     = f2bf(tr * c - ti * sn);
    out[ob + 1] = f2bf(tr * sn + ti * c);
}

__global__ void k_rope_kv(const float* __restrict__ kv, const float* __restrict__ cosb,
                          const float* __restrict__ sinb,
                          u16* __restrict__ kout, u16* __restrict__ vout)
{
    int idx = blockIdx.x * blockDim.x + threadIdx.x;       // B*S*HKV*64 threads
    int j = idx & 63; int t = idx >> 6;
    int hk = t & 3;   t >>= 2;
    int s  = t & 2047; int b = t >> 11;
    size_t base = ((size_t)(b * S_SZ + s)) * (2 * HKV_N * DK_N) + hk * DK_N + 2 * j;
    float tr = kv[base], ti = kv[base + 1];
    float c = cosb[s * 64 + j], sn = sinb[s * 64 + j];
    size_t ob = (((size_t)(b * HKV_N + hk)) * S_SZ + s) * DK_N + 2 * j;
    kout[ob]     = f2bf(tr * c - ti * sn);
    kout[ob + 1] = f2bf(tr * sn + ti * c);
    vout[ob]     = f2bf(kv[base + 512]);                   // v = kv[:,:,1]
    vout[ob + 1] = f2bf(kv[base + 512 + 1]);
}

// -------------------------------------------------------- flash attention ----
// grid: B*HQ * (S/64) blocks, 128 threads (4 waves).  Wave w owns q rows
// [tile*64 + w*16, +16) for one head; streams 32-key tiles causally.
__global__ __launch_bounds__(128) void k_attn(
    const u16* __restrict__ Q, const u16* __restrict__ Kh,
    const u16* __restrict__ Vh, u16* __restrict__ ctx)
{
    __shared__ u16 vT[4][128 * 40];    // per-wave V^T tile [d][key], padded
    __shared__ u16 pS[4][16 * 32];     // per-wave P tile   [row][key]
    const int wave = threadIdx.x >> 5, lane = threadIdx.x & 31;
    const int hb = lane >> 4, l15 = lane & 15;

    const int bh = blockIdx.x >> 5;            // S/64 == 32 q-tiles
    const int tile = blockIdx.x & 31;
    const int b = bh >> 4, hq = bh & 15, hk = hq >> 2;
    const int qt = tile * 64 + wave * 16;

    const u16* qbase = Q  + (((size_t)(b * HQ_N  + hq)) * S_SZ + qt) * DK_N;
    const u16* kbase = Kh + (((size_t)(b * HKV_N + hk)) * S_SZ) * DK_N;
    const u16* vbase = Vh + (((size_t)(b * HKV_N + hk)) * S_SZ) * DK_N;
    u16* vtw = vT[wave];
    u16* psw = pS[wave];

    // preload Q fragments (DK=128 -> 4 K-slices of 32)
    bf16x16 qf[4];
    {
        const u16* qrow = qbase + (size_t)l15 * DK_N + hb * 8;
        #pragma unroll
        for (int kk = 0; kk < 4; ++kk)
            qf[kk] = make_frag(qrow + kk * 32, qrow + kk * 32 + 16);
    }

    float m[8], l[8];
    f32x8 o[8];
    #pragma unroll
    for (int i = 0; i < 8; ++i) {
        m[i] = -3.0e38f; l[i] = 0.0f;
        #pragma unroll
        for (int sl = 0; sl < 8; ++sl) o[sl][i] = 0.0f;
    }
    const float scale = 0.08838834764831845f;   // 1/sqrt(128)

    for (int k0 = 0; k0 < qt + 16; k0 += 32) {
        // stage V^T into LDS: lane = key row
        {
            const u16* vrow = vbase + (size_t)(k0 + lane) * DK_N;
            #pragma unroll
            for (int d8 = 0; d8 < 16; ++d8) {
                uint4 u = *(const uint4*)(vrow + d8 * 8);
                unsigned int ww[4] = {u.x, u.y, u.z, u.w};
                #pragma unroll
                for (int e = 0; e < 4; ++e) {
                    int d = d8 * 8 + e * 2;
                    vtw[(d    ) * 40 + lane] = (u16)(ww[e] & 0xFFFFu);
                    vtw[(d + 1) * 40 + lane] = (u16)(ww[e] >> 16);
                }
            }
        }
        asm volatile("" ::: "memory");

        // scores: two 16x16 tiles (keys k0..k0+15, k0+16..k0+31)
        f32x8 s0, s1;
        #pragma unroll
        for (int i = 0; i < 8; ++i) { s0[i] = 0.0f; s1[i] = 0.0f; }
        #pragma unroll
        for (int t = 0; t < 2; ++t) {
            const u16* krow = kbase + (size_t)(k0 + t * 16 + l15) * DK_N + hb * 8;
            #pragma unroll
            for (int kk = 0; kk < 4; ++kk) {
                bf16x16 bf = make_frag(krow + kk * 32, krow + kk * 32 + 16);
                if (t == 0)
                    s0 = __builtin_amdgcn_wmma_f32_16x16x32_bf16(
                        false, qf[kk], false, bf, (short)0, s0, false, false);
                else
                    s1 = __builtin_amdgcn_wmma_f32_16x16x32_bf16(
                        false, qf[kk], false, bf, (short)0, s1, false, false);
            }
        }

        // online softmax update (rows live in 16-lane halves)
        #pragma unroll
        for (int i = 0; i < 8; ++i) {
            int qrow = qt + i + hb * 8;
            int key0 = k0 + l15, key1 = key0 + 16;
            float v0 = s0[i] * scale; if (key0 > qrow) v0 = -3.0e38f;
            float v1 = s1[i] * scale; if (key1 > qrow) v1 = -3.0e38f;
            float rm = fmaxf(v0, v1);
            #pragma unroll
            for (int off = 1; off < 16; off <<= 1)
                rm = fmaxf(rm, __shfl_xor(rm, off, 16));
            float mn   = fmaxf(m[i], rm);
            float corr = __expf(m[i] - mn);
            float p0 = __expf(v0 - mn), p1 = __expf(v1 - mn);
            float rs = p0 + p1;
            #pragma unroll
            for (int off = 1; off < 16; off <<= 1)
                rs += __shfl_xor(rs, off, 16);
            l[i] = l[i] * corr + rs;
            m[i] = mn;
            #pragma unroll
            for (int sl = 0; sl < 8; ++sl) o[sl][i] *= corr;
            int r = i + hb * 8;
            psw[r * 32 + l15]      = f2bf(p0);
            psw[r * 32 + l15 + 16] = f2bf(p1);
        }
        asm volatile("" ::: "memory");

        // P (A-frag) from LDS, then P @ V accumulation over 8 d-slices
        const u16* pp = psw + l15 * 32 + hb * 8;
        bf16x16 pa = make_frag(pp, pp + 16);
        #pragma unroll
        for (int sl = 0; sl < 8; ++sl) {
            const u16* vp = vtw + (sl * 16 + l15) * 40 + hb * 8;
            bf16x16 bf = make_frag(vp, vp + 16);
            o[sl] = __builtin_amdgcn_wmma_f32_16x16x32_bf16(
                false, pa, false, bf, (short)0, o[sl], false, false);
        }
        asm volatile("" ::: "memory");
    }

    // normalize and write ctx (bf16, [b][s][hq*128+d] row-major)
    #pragma unroll
    for (int sl = 0; sl < 8; ++sl) {
        #pragma unroll
        for (int i = 0; i < 8; ++i) {
            int r = qt + i + hb * 8;
            float val = o[sl][i] / l[i];
            ctx[((size_t)(b * S_SZ + r)) * DM + hq * DK_N + sl * 16 + l15] = f2bf(val);
        }
    }
}

// ------------------------------------------------------------------ launch ---
extern "C" void kernel_launch(void* const* d_in, const int* in_sizes, int n_in,
                              void* d_out, int out_size, void* d_ws, size_t ws_size,
                              hipStream_t stream) {
    const float* x      = (const float*)d_in[0];
    const float* w_q    = (const float*)d_in[1];
    const float* w_kv   = (const float*)d_in[2];
    const float* fc_w   = (const float*)d_in[3];
    const float* fc_b   = (const float*)d_in[4];
    const float* f_cos  = (const float*)d_in[5];
    const float* f_sin  = (const float*)d_in[6];
    float* out = (float*)d_out;

    const size_t M  = (size_t)B_SZ * S_SZ;       // 4096
    char* wsb = (char*)d_ws;
    size_t off = 0;
    auto carve = [&](size_t bytes) -> void* {
        void* p = wsb + off;
        off = (off + bytes + 255) & ~(size_t)255;
        return p;
    };
    u16*   xbf   = (u16*)  carve(M * DM * 2);
    u16*   wqT   = (u16*)  carve((size_t)DM * DM * 2);     // [n][k]
    u16*   wkvT  = (u16*)  carve((size_t)1024 * DM * 2);   // [n][k]
    u16*   fcwT  = (u16*)  carve((size_t)DM * DM * 2);     // [n][k]
    float* qf    = (float*)carve(M * DM * 4);
    float* kvf   = (float*)carve(M * 1024 * 4);
    u16*   qbf   = (u16*)  carve((size_t)B_SZ * HQ_N  * S_SZ * DK_N * 2);
    u16*   kbf   = (u16*)  carve((size_t)B_SZ * HKV_N * S_SZ * DK_N * 2);
    u16*   vbf   = (u16*)  carve((size_t)B_SZ * HKV_N * S_SZ * DK_N * 2);
    u16*   ctxbf = (u16*)  carve(M * DM * 2);
    (void)ws_size; (void)n_in; (void)in_sizes; (void)out_size;

    // 1) fp32 -> bf16 conversions (weights transposed to [n][k])
    k_f32_to_bf16<<<2048, 256, 0, stream>>>(x, xbf, (int)(M * DM));
    k_f32_to_bf16_T<<<dim3(DM / 32, DM / 32),   dim3(32, 8), 0, stream>>>(w_q,  wqT,  DM, DM);
    k_f32_to_bf16_T<<<dim3(1024 / 32, DM / 32), dim3(32, 8), 0, stream>>>(w_kv, wkvT, DM, 1024);
    k_f32_to_bf16_T<<<dim3(DM / 32, DM / 32),   dim3(32, 8), 0, stream>>>(fc_w, fcwT, DM, DM);

    // 2) projections (fp32 accumulate, fp32 out for RoPE precision)
    k_gemm_bf16<<<dim3(32, 32), 128, 0, stream>>>(xbf, wqT,  qf,  nullptr, (int)M, DM,   DM);
    k_gemm_bf16<<<dim3(32, 16), 128, 0, stream>>>(xbf, wkvT, kvf, nullptr, (int)M, 1024, DM);

    // 3) RoPE + head-major repack to bf16
    k_rope_q <<<16384, 256, 0, stream>>>(qf,  f_cos, f_sin, qbf);
    k_rope_kv<<< 4096, 256, 0, stream>>>(kvf, f_cos, f_sin, kbf, vbf);

    // 4) causal GQA flash attention
    k_attn<<<B_SZ * HQ_N * (S_SZ / 64), 128, 0, stream>>>(qbf, kbf, vbf, ctxbf);

    // 5) output projection with fused bias -> fp32 d_out
    k_gemm_bf16<<<dim3(32, 32), 128, 0, stream>>>(ctxbf, fcwT, out, fc_b, (int)M, DM, DM);
}
